// MultiEmbodimentActionEncoder_2723009266341
// MI455X (gfx1250) — matrix operand order repaired and checked by reading, not compiled, not run on hardware
//
#include <hip/hip_runtime.h>
#include <hip/hip_bf16.h>
#include <cstdint>

// ---- CDNA5 WMMA types --------------------------------------------------
typedef __attribute__((ext_vector_type(16))) __bf16 v16bf;
typedef __attribute__((ext_vector_type(8)))  float  v8f;

union FragU {
    uint4  q[2];
    v16bf  v;
};

// native f32 -> bf16 convert (single v_cvt on gfx1250)
__device__ __forceinline__ unsigned short f2bf(float f) {
    __bf16 h = (__bf16)f;
    return __builtin_bit_cast(unsigned short, h);
}

// A-fragment (16x32 bf16, MxK) from LDS, per ISA 16-bit A layout:
//   lanes 0-15 (g=0): elems 0..7 = K 0..7, elems 8..15 = K 16..23
//   lanes 16-31(g=1): elems 0..7 = K 8..15, elems 8..15 = K 24..31
__device__ __forceinline__ v16bf load_afrag(const unsigned short* base,
                                            int stride_h, int m, int g, int kofs) {
    const unsigned short* p = base + m * stride_h + kofs + g * 8;
    FragU f;
    f.q[0] = *(const uint4*)(p);        // K = kofs + g*8 .. +7
    f.q[1] = *(const uint4*)(p + 16);   // K = kofs + 16 + g*8 .. +7
    return f.v;
}

// B-fragment (32x16 bf16, KxN) from pre-shuffled global weights:
// each lane's 16 halfs are stored contiguously (32 B per lane, 1 KB per tile).
__device__ __forceinline__ v16bf load_bfrag(const unsigned short* tileBase, int lane) {
    const unsigned short* p = tileBase + lane * 16;
    FragU f;
    f.q[0] = *(const uint4*)(p);
    f.q[1] = *(const uint4*)(p + 8);
    return f.v;
}

#define WMMA_BF16(a, b, c) \
    __builtin_amdgcn_wmma_f32_16x16x32_bf16(false, (a), false, (b), (short)0, (c), false, false)

// ---- prep: convert f32 weights -> bf16, fragment-major tiles -----------
// dst layout: [E][K/32][N/16][lane 0..31][half 0..15]
// For lane (n = lane&15, g = lane>>4), half h: element = W[kt*32 + (h>>3)*16 + g*8 + (h&7)][nt*16 + n]
__global__ void shuffle_w_kernel(const float* __restrict__ src,
                                 unsigned short* __restrict__ dst,
                                 int K, int N) {
    size_t d = (size_t)blockIdx.x * 256 + threadIdx.x;
    int KT = K >> 5, NT = N >> 4;
    size_t perE = (size_t)KT * NT * 512;
    int    e    = (int)(d / perE);
    size_t rem  = d % perE;
    int tile = (int)(rem >> 9);
    int elem = (int)(rem & 511);
    int kt = tile / NT, nt = tile % NT;
    int lane = elem >> 4, h = elem & 15;
    int n = lane & 15, g = lane >> 4;
    int kl = (h >> 3) * 16 + g * 8 + (h & 7);
    float x = src[(size_t)e * K * N + (size_t)(kt * 32 + kl) * N + nt * 16 + n];
    dst[d] = f2bf(x);
}

// ---- main kernel: one workgroup (8 waves) per sample -------------------
// wave w: row-tile rt = w & 3 (rows rt*16..), col half ch = w >> 2 (cols ch*256..)
#define STR   520   // LDS half-stride: 1040 B = 16*65 (b128-aligned, bank-spread)
#define ASTR  40    // actions stride: 80 B = 16*5

__global__ void __launch_bounds__(256)
mea_encoder_kernel(const float* __restrict__ actions,
                   const float* __restrict__ timesteps,
                   const int*   __restrict__ cat_ids,
                   const float* __restrict__ b1,
                   const float* __restrict__ b2,
                   const float* __restrict__ b3,
                   const unsigned short* __restrict__ wsW1,
                   const unsigned short* __restrict__ wsW2,
                   const unsigned short* __restrict__ wsW3,
                   float* __restrict__ out) {
    extern __shared__ __align__(16) unsigned char smem[];
    unsigned short* bufA = (unsigned short*)smem;        // [4][16][STR] a_emb (bf16)
    unsigned short* bufB = bufA + 4 * 16 * STR;          // [4][16][STR] swish(h)
    unsigned short* actS = bufB + 4 * 16 * STR;          // [64][ASTR] actions (bf16)
    unsigned short* tauS = actS + 64 * ASTR;             // [512] tau posenc (bf16)
    float*          tvec = (float*)(tauS + 512);         // [512] tau@W2_bot + b2

    const int b    = blockIdx.x;
    const int tid  = threadIdx.x;
    const int lane = tid & 31;
    const int wave = tid >> 5;
    const int ml   = lane & 15;      // A-frag row / D-frag col
    const int g    = lane >> 4;
    const int rt   = wave & 3;       // row tile 0..3
    const int ch   = wave >> 2;      // col half 0..1
    const int cat  = cat_ids[b];

    // ---- stage actions (64x32 f32 -> bf16 LDS) + sinusoidal posenc ----
    {
        const float* asrc = actions + (size_t)b * 64 * 32;
        #pragma unroll
        for (int i = 0; i < 8; ++i) {
            int idx = tid + i * 256;
            actS[(idx >> 5) * ASTR + (idx & 31)] = f2bf(asrc[idx]);
        }
        float t   = timesteps[b];
        float ang = t * __expf(-(float)tid * 0.036118982f);  // ln(1e4)/255
        tauS[tid]       = f2bf(__sinf(ang));
        tauS[256 + tid] = f2bf(__cosf(ang));
    }
    __syncthreads();

    const unsigned short* w1e = wsW1 + (size_t)cat * (32 * 512);
    const unsigned short* w2e = wsW2 + (size_t)cat * (32 * 32 * 512);
    const unsigned short* w3e = wsW3 + (size_t)cat * (16 * 32 * 512);

    // ---- layer 1: a_emb = actions @ W1 + b1 (K=32, one WMMA per tile) ----
    {
        v16bf aAct = load_afrag(actS + rt * 16 * ASTR, ASTR, ml, g, 0);
        const float* b1e = b1 + cat * 512;
        #pragma unroll
        for (int t = 0; t < 16; ++t) {
            int nt = ch * 16 + t;
            float bv = b1e[nt * 16 + ml];
            v8f c;
            #pragma unroll
            for (int r = 0; r < 8; ++r) c[r] = bv;
            v16bf bf = load_bfrag(w1e + (size_t)nt * 512, lane);
            c = WMMA_BF16(aAct, bf, c);
            #pragma unroll
            for (int r = 0; r < 8; ++r)
                bufA[rt * 16 * STR + (g * 8 + r) * STR + nt * 16 + ml] = f2bf(c[r]);
        }
    }

    // ---- tvec = tau @ W2[512:1024] + b2  (rank-1 term; 4 col-tiles/wave) ----
    {
        const float* b2e = b2 + cat * 512;
        #pragma unroll
        for (int j = 0; j < 4; ++j) {
            int ct = wave * 4 + j;
            float bv = b2e[ct * 16 + ml];
            v8f c;
            #pragma unroll
            for (int r = 0; r < 8; ++r) c[r] = bv;
            #pragma unroll 1
            for (int k = 0; k < 16; ++k) {
                v16bf a  = load_afrag(tauS, 0, 0, g, k * 32);   // rows all equal
                v16bf bf = load_bfrag(w2e + (size_t)((16 + k) * 32 + ct) * 512, lane);
                c = WMMA_BF16(a, bf, c);
            }
            if (g == 0) tvec[ct * 16 + ml] = c[0];   // row 0 of tile
        }
    }
    __syncthreads();

    // ---- layer 2: h = a_emb @ W2[:512] + tvec ; x = swish(h) ----
    v8f acc[16];
    #pragma unroll
    for (int t = 0; t < 16; ++t) {
        float bv = tvec[(ch * 16 + t) * 16 + ml];
        #pragma unroll
        for (int r = 0; r < 8; ++r) acc[t][r] = bv;
    }
    #pragma unroll 1
    for (int k = 0; k < 16; ++k) {
        v16bf a = load_afrag(bufA + rt * 16 * STR, STR, ml, g, k * 32);
        #pragma unroll
        for (int t = 0; t < 16; ++t) {
            v16bf bf = load_bfrag(w2e + (size_t)(k * 32 + ch * 16 + t) * 512, lane);
            acc[t] = WMMA_BF16(a, bf, acc[t]);
        }
    }
    #pragma unroll
    for (int t = 0; t < 16; ++t) {
        #pragma unroll
        for (int r = 0; r < 8; ++r) {
            float h = acc[t][r];
            // swish via fast reciprocal: h * rcp(1 + exp(-h))
            float s = h * __builtin_amdgcn_rcpf(1.f + __expf(-h));
            bufB[rt * 16 * STR + (g * 8 + r) * STR + (ch * 16 + t) * 16 + ml] = f2bf(s);
        }
    }
    __syncthreads();

    // ---- layer 3: out = x @ W3 + b3 ----
    {
        const float* b3e = b3 + cat * 512;
        #pragma unroll
        for (int t = 0; t < 16; ++t) {
            float bv = b3e[(ch * 16 + t) * 16 + ml];
            #pragma unroll
            for (int r = 0; r < 8; ++r) acc[t][r] = bv;
        }
    }
    #pragma unroll 1
    for (int k = 0; k < 16; ++k) {
        v16bf a = load_afrag(bufB + rt * 16 * STR, STR, ml, g, k * 32);
        #pragma unroll
        for (int t = 0; t < 16; ++t) {
            v16bf bf = load_bfrag(w3e + (size_t)(k * 32 + ch * 16 + t) * 512, lane);
            acc[t] = WMMA_BF16(a, bf, acc[t]);
        }
    }
    float* op = out + (size_t)b * 64 * 512;
    #pragma unroll
    for (int t = 0; t < 16; ++t) {
        #pragma unroll
        for (int r = 0; r < 8; ++r)
            op[(size_t)(rt * 16 + g * 8 + r) * 512 + (ch * 16 + t) * 16 + ml] = acc[t][r];
    }
}

// ---- host launcher ------------------------------------------------------
extern "C" void kernel_launch(void* const* d_in, const int* in_sizes, int n_in,
                              void* d_out, int out_size, void* d_ws, size_t ws_size,
                              hipStream_t stream) {
    const float* actions   = (const float*)d_in[0];
    const float* timesteps = (const float*)d_in[1];
    const int*   cat_ids   = (const int*)d_in[2];
    const float* W1        = (const float*)d_in[3];
    const float* b1        = (const float*)d_in[4];
    const float* W2        = (const float*)d_in[5];
    const float* b2        = (const float*)d_in[6];
    const float* W3        = (const float*)d_in[7];
    const float* b3        = (const float*)d_in[8];
    float*       out       = (float*)d_out;

    // workspace: bf16 fragment-major weights
    unsigned short* wsW1 = (unsigned short*)d_ws;                  // 16*32*512      = 262,144 halfs
    unsigned short* wsW2 = wsW1 + (size_t)16 * 32 * 512;           // 16*32*32*512   = 8,388,608
    unsigned short* wsW3 = wsW2 + (size_t)16 * 32 * 32 * 512;      // 16*16*32*512   = 4,194,304

    shuffle_w_kernel<<<(16 * 32 * 512) / 256,        256, 0, stream>>>(W1, wsW1, 32,   512);
    shuffle_w_kernel<<<(16 * 32 * 32 * 512) / 256,   256, 0, stream>>>(W2, wsW2, 1024, 512);
    shuffle_w_kernel<<<(16 * 16 * 32 * 512) / 256,   256, 0, stream>>>(W3, wsW3, 512,  512);

    size_t smem = (size_t)(4 * 16 * STR * 2) * 2   // bufA + bufB
                + (size_t)(64 * ASTR * 2)          // actions
                + (size_t)(512 * 2)                // tau
                + (size_t)(512 * 4);               // tvec
    mea_encoder_kernel<<<256, 256, smem, stream>>>(
        actions, timesteps, cat_ids, b1, b2, b3, wsW1, wsW2, wsW3, out);
}
